// MultiHeadCrossAttention_50757923504657
// MI455X (gfx1250) — matrix operand order repaired and checked by reading, not compiled, not run on hardware
//
#include <hip/hip_runtime.h>
#include <hip/hip_bf16.h>

#define DEV __device__ __forceinline__
#define USE_ASYNC 1

typedef __attribute__((ext_vector_type(16))) __bf16 v16bf;
typedef __attribute__((ext_vector_type(8)))  float  v8f;

constexpr int Bb = 4, Ss = 2048, Dd = 1024, Hh = 16, HD = 64;

union AFrag { v16bf bf; unsigned int w[8]; };

DEV unsigned short f2bf(float f) {
    union { float f; unsigned u; } v; v.f = f;
    unsigned r = v.u + 0x7FFFu + ((v.u >> 16) & 1u);   // round-to-nearest-even
    return (unsigned short)(r >> 16);
}

// ---- CDNA5 async global->LDS copy (16 bytes per lane), tracked by ASYNCcnt ----
DEV void async_ld16(void* lds, const void* gptr) {
#if USE_ASYNC
    unsigned l = (unsigned)(unsigned long long)lds;            // LDS byte offset (flat addr truncation)
    unsigned long long g = (unsigned long long)gptr;
    asm volatile("global_load_async_to_lds_b128 %0, %1, off"
                 :: "v"(l), "v"(g) : "memory");
#else
    *(uint4*)lds = *(const uint4*)gptr;
#endif
}
DEV void async_wait() {
#if USE_ASYNC
    asm volatile("s_wait_asynccnt 0x0" ::: "memory");
#endif
}

// ---------------- fp32 -> bf16 conversion ----------------
__global__ void cvt4_kernel(const float4* __restrict__ in,
                            ushort4* __restrict__ out, int n4) {
    int i  = blockIdx.x * blockDim.x + threadIdx.x;
    int st = gridDim.x * blockDim.x;
    for (; i < n4; i += st) {
        float4 f = in[i];
        ushort4 o;
        o.x = f2bf(f.x); o.y = f2bf(f.y); o.z = f2bf(f.z); o.w = f2bf(f.w);
        out[i] = o;
    }
}

// ---------------- generic bf16 WMMA GEMM ----------------
// C = A(MxK,bf16,row) @ B(KxN,bf16,row) + bias
// MODE 0: fp32 row-major to outF
// MODE 1: kv split -> outK/outV as [B,H,S,HD] bf16  (N = 2*D)
// MODE 2: q        -> outK as [B,H,S,HD] bf16       (N = D)
template <int MODE>
__global__ __launch_bounds__(256) void gemm_bf16_kernel(
    const unsigned short* __restrict__ A, const unsigned short* __restrict__ Bm,
    const float* __restrict__ bias,
    float* __restrict__ outF, unsigned short* __restrict__ outK,
    unsigned short* __restrict__ outV, int M, int N, int K) {
    __shared__ unsigned short As[128 * 32];   // [m][k]
    __shared__ unsigned short Bs[128 * 32];   // [n][k] (transposed stage)

    const int tid  = threadIdx.x;
    const int wave = tid >> 5, lane = tid & 31;
    const int half = lane >> 4, l16 = lane & 15;
    const int wm = wave >> 1, wn = wave & 1;          // 4x2 wave grid
    const int m0 = blockIdx.y * 128;
    const int n0 = blockIdx.x * 128;

    v8f acc[2][4];
#pragma unroll
    for (int i = 0; i < 2; i++)
#pragma unroll
        for (int j = 0; j < 4; j++) acc[i][j] = (v8f){0.f,0.f,0.f,0.f,0.f,0.f,0.f,0.f};

    const int ar = tid >> 1;                 // A-tile row this thread stages
    const int ac = (tid & 1) * 16;           // A-tile col group
    const int bk = tid >> 3;                 // B-tile k this thread stages
    const int bc = (tid & 7) * 16;           // B-tile col group

    for (int k0 = 0; k0 < K; k0 += 32) {
        __syncthreads();
        // A tile 128x32: async straight copy, 2x b128 per thread
        async_ld16(As + ar * 32 + ac,     A + (size_t)(m0 + ar) * K + k0 + ac);
        async_ld16(As + ar * 32 + ac + 8, A + (size_t)(m0 + ar) * K + k0 + ac + 8);
        // B tile 32x128 -> Bs[n][k] transposed (manual: needs transpose)
        {
            const unsigned short* src = Bm + (size_t)(k0 + bk) * N + n0 + bc;
            if (k0 + 32 < K)
                __builtin_prefetch(src + (size_t)32 * N, 0, 1);   // warm next tile
#pragma unroll
            for (int i = 0; i < 16; i++) Bs[(bc + i) * 32 + bk] = src[i];
        }
        async_wait();
        __syncthreads();

        AFrag a[2], b[4];
#pragma unroll
        for (int i = 0; i < 2; i++) {
            int row = wm * 32 + i * 16 + l16;
#pragma unroll
            for (int j = 0; j < 8; j++) {
                int kk = (j < 4) ? (2 * j + 8 * half) : (16 + 2 * (j - 4) + 8 * half);
                a[i].w[j] = *(const unsigned int*)(As + row * 32 + kk);
            }
        }
#pragma unroll
        for (int j4 = 0; j4 < 4; j4++) {
            int col = wn * 64 + j4 * 16 + l16;
#pragma unroll
            for (int j = 0; j < 8; j++) {
                int kk = (j < 4) ? (2 * j + 8 * half) : (16 + 2 * (j - 4) + 8 * half);
                b[j4].w[j] = *(const unsigned int*)(Bs + col * 32 + kk);
            }
        }
#pragma unroll
        for (int i = 0; i < 2; i++)
#pragma unroll
            for (int j = 0; j < 4; j++)
                acc[i][j] = __builtin_amdgcn_wmma_f32_16x16x32_bf16(
                    false, a[i].bf, false, b[j].bf, (short)0, acc[i][j], false, false);
    }

#pragma unroll
    for (int i = 0; i < 2; i++) {
        int rowb = m0 + wm * 32 + i * 16;
#pragma unroll
        for (int j4 = 0; j4 < 4; j4++) {
            int col = n0 + wn * 64 + j4 * 16 + l16;
            float bv = bias[col];
#pragma unroll
            for (int j = 0; j < 8; j++) {
                int r = rowb + j + 8 * half;
                float val = acc[i][j4][j] + bv;
                if (MODE == 0) {
                    outF[(size_t)r * N + col] = val;
                } else if (MODE == 1) {
                    int bi = r >> 11, s = r & (Ss - 1);
                    int h = col >> 7, inner = col & 127;
                    size_t idx = (((size_t)bi * Hh + h) * Ss + s) * HD + (inner & 63);
                    unsigned short bfv = f2bf(val);
                    if (inner < 64) outK[idx] = bfv; else outV[idx] = bfv;
                } else {
                    int bi = r >> 11, s = r & (Ss - 1);
                    int h = col >> 6, hd = col & 63;
                    outK[(((size_t)bi * Hh + h) * Ss + s) * HD + hd] = f2bf(val);
                }
            }
        }
    }
}

// ---------------- flash attention (bf16 WMMA, f32 accum) ----------------
// grid: (S/64, B*H), block 128 (4 waves); wave w handles 16 query rows
__global__ __launch_bounds__(128) void attn_kernel(
    const unsigned short* __restrict__ Q,   // [B,H,S,HD]
    const unsigned short* __restrict__ Kb,  // [B,H,S,HD]
    const unsigned short* __restrict__ Vb,  // [B,H,S,HD]
    const float* __restrict__ mask,         // [S,S]
    unsigned short* __restrict__ ctx) {     // [B,S,D] bf16
    __shared__ unsigned short Kt[64 * 64];      // [key][hd]  (B-operand [N][K] for QK^T)
    __shared__ unsigned short Vt[64 * 64];      // [hd][key]  (B-operand [N][K] for P@V)
    __shared__ unsigned short Ps[4 * 16 * 64];  // per-wave P staging / Q staging

    const int tid  = threadIdx.x;
    const int wave = tid >> 5, lane = tid & 31;
    const int half = lane >> 4, l16 = lane & 15;
    const int bh = blockIdx.y;
    const int b = bh >> 4, h = bh & 15;
    const int q0 = blockIdx.x * 64;

    const size_t headOff = (size_t)bh * Ss * HD;
    const unsigned short* Qh = Q + headOff;
    const unsigned short* Kh = Kb + headOff;
    const unsigned short* Vh = Vb + headOff;

    const int tr = tid >> 1;                 // staged row (0..63)
    const int tc = (tid & 1) * 32;           // staged col group (0/32)

    // stage Q tile (64x64) into Ps via async copy, build A fragments (K=64 -> 2 frags)
#pragma unroll
    for (int i = 0; i < 4; i++)
        async_ld16(Ps + tr * 64 + tc + i * 8, Qh + (size_t)(q0 + tr) * HD + tc + i * 8);
    async_wait();
    __syncthreads();

    AFrag qf[2];
#pragma unroll
    for (int kb = 0; kb < 2; kb++) {
        int row = wave * 16 + l16;
#pragma unroll
        for (int j = 0; j < 8; j++) {
            int kk = kb * 32 + ((j < 4) ? (2 * j + 8 * half) : (16 + 2 * (j - 4) + 8 * half));
            qf[kb].w[j] = *(const unsigned int*)(Ps + row * 64 + kk);
        }
    }

    v8f O[4];
#pragma unroll
    for (int j = 0; j < 4; j++) O[j] = (v8f){0.f,0.f,0.f,0.f,0.f,0.f,0.f,0.f};
    float mrow[8], lrow[8];
#pragma unroll
    for (int j = 0; j < 8; j++) { mrow[j] = -1e30f; lrow[j] = 0.f; }
    const float scale = 0.125f;          // 1/sqrt(64)
    const float log2e = 1.44269504f;

    for (int kt = 0; kt < Ss; kt += 64) {
        __syncthreads();   // previous iteration finished reading Kt/Vt/Ps
        // K tile -> Kt[key][hd]: async straight copy (4x b128 per thread)
#pragma unroll
        for (int i = 0; i < 4; i++)
            async_ld16(Kt + tr * 64 + tc + i * 8, Kh + (size_t)(kt + tr) * HD + tc + i * 8);
        // V tile transposed -> Vt[hd][key] (manual: needs transpose)
        {
            union { uint4 u4[4]; unsigned short us[32]; } tmp;
            const uint4* src = (const uint4*)(Vh + (size_t)(kt + tr) * HD + tc);
            tmp.u4[0] = src[0]; tmp.u4[1] = src[1]; tmp.u4[2] = src[2]; tmp.u4[3] = src[3];
#pragma unroll
            for (int i = 0; i < 32; i++) Vt[(tc + i) * 64 + tr] = tmp.us[i];
        }
        async_wait();
        __syncthreads();

        // S = Q @ K^T  (16q x 64keys per wave)
        v8f Sa[4];
#pragma unroll
        for (int j = 0; j < 4; j++) Sa[j] = (v8f){0.f,0.f,0.f,0.f,0.f,0.f,0.f,0.f};
#pragma unroll
        for (int n4 = 0; n4 < 4; n4++) {
            int key = n4 * 16 + l16;
#pragma unroll
            for (int kb = 0; kb < 2; kb++) {
                AFrag bfr;
#pragma unroll
                for (int j = 0; j < 8; j++) {
                    int kk = kb * 32 + ((j < 4) ? (2 * j + 8 * half) : (16 + 2 * (j - 4) + 8 * half));
                    bfr.w[j] = *(const unsigned int*)(Kt + key * 64 + kk);
                }
                Sa[n4] = __builtin_amdgcn_wmma_f32_16x16x32_bf16(
                    false, qf[kb].bf, false, bfr.bf, (short)0, Sa[n4], false, false);
            }
        }

        // scale + mask + online softmax (rows live across 16-lane halves)
        float mnew[8];
#pragma unroll
        for (int j = 0; j < 8; j++) {
            int qr = q0 + wave * 16 + j + 8 * half;
            float mx = -1e30f;
#pragma unroll
            for (int n4 = 0; n4 < 4; n4++) {
                int key = kt + n4 * 16 + l16;
                float s = Sa[n4][j] * scale + mask[(size_t)qr * Ss + key];
                Sa[n4][j] = s;
                mx = fmaxf(mx, s);
            }
            mx = fmaxf(mx, __shfl_xor(mx, 1, 32));
            mx = fmaxf(mx, __shfl_xor(mx, 2, 32));
            mx = fmaxf(mx, __shfl_xor(mx, 4, 32));
            mx = fmaxf(mx, __shfl_xor(mx, 8, 32));
            mnew[j] = fmaxf(mrow[j], mx);
        }
#pragma unroll
        for (int j = 0; j < 8; j++) {
            float alpha = exp2f((mrow[j] - mnew[j]) * log2e);
            float rs = 0.f;
#pragma unroll
            for (int n4 = 0; n4 < 4; n4++) {
                float p = exp2f((Sa[n4][j] - mnew[j]) * log2e);
                Sa[n4][j] = p;
                rs += p;
            }
            rs += __shfl_xor(rs, 1, 32);
            rs += __shfl_xor(rs, 2, 32);
            rs += __shfl_xor(rs, 4, 32);
            rs += __shfl_xor(rs, 8, 32);
            lrow[j] = lrow[j] * alpha + rs;
            mrow[j] = mnew[j];
#pragma unroll
            for (int n4 = 0; n4 < 4; n4++) O[n4][j] *= alpha;
        }

        // P: C-layout -> bf16 in LDS -> A-fragment layout
        unsigned short* Pw = Ps + wave * 16 * 64;
#pragma unroll
        for (int n4 = 0; n4 < 4; n4++)
#pragma unroll
            for (int j = 0; j < 8; j++)
                Pw[(j + 8 * half) * 64 + n4 * 16 + l16] = f2bf(Sa[n4][j]);
        __syncthreads();

        AFrag pf[2];
#pragma unroll
        for (int kb = 0; kb < 2; kb++) {
            int row = l16;
#pragma unroll
            for (int j = 0; j < 8; j++) {
                int kk = kb * 32 + ((j < 4) ? (2 * j + 8 * half) : (16 + 2 * (j - 4) + 8 * half));
                pf[kb].w[j] = *(const unsigned int*)(Pw + row * 64 + kk);
            }
        }
#pragma unroll
        for (int n4 = 0; n4 < 4; n4++) {
            int hd = n4 * 16 + l16;
#pragma unroll
            for (int kb = 0; kb < 2; kb++) {
                AFrag bvr;
#pragma unroll
                for (int j = 0; j < 8; j++) {
                    int kk = kb * 32 + ((j < 4) ? (2 * j + 8 * half) : (16 + 2 * (j - 4) + 8 * half));
                    bvr.w[j] = *(const unsigned int*)(Vt + hd * 64 + kk);
                }
                O[n4] = __builtin_amdgcn_wmma_f32_16x16x32_bf16(
                    false, pf[kb].bf, false, bvr.bf, (short)0, O[n4], false, false);
            }
        }
    }

    // finalize: O /= l, write context bf16 [B,S,D]
#pragma unroll
    for (int j = 0; j < 8; j++) {
        float inv = 1.f / lrow[j];
        int s = q0 + wave * 16 + j + 8 * half;
#pragma unroll
        for (int n4 = 0; n4 < 4; n4++) {
            int hd = n4 * 16 + l16;
            ctx[((size_t)b * Ss + s) * Dd + h * HD + hd] = f2bf(O[n4][j] * inv);
        }
    }
}

// ---------------- launcher ----------------
extern "C" void kernel_launch(void* const* d_in, const int* in_sizes, int n_in,
                              void* d_out, int out_size, void* d_ws, size_t ws_size,
                              hipStream_t stream) {
    (void)in_sizes; (void)n_in; (void)out_size; (void)ws_size;
    const float* x    = (const float*)d_in[0];
    const float* y    = (const float*)d_in[1];
    const float* mask = (const float*)d_in[2];
    const float* Wkv  = (const float*)d_in[3];
    const float* bkv  = (const float*)d_in[4];
    const float* Wq   = (const float*)d_in[5];
    const float* bq   = (const float*)d_in[6];
    const float* Wo   = (const float*)d_in[7];
    const float* bo   = (const float*)d_in[8];

    unsigned short* ws = (unsigned short*)d_ws;
    size_t off = 0;
    auto alloc = [&](size_t n) { unsigned short* p = ws + off; off += n; return p; };
    const size_t ND = (size_t)Bb * Ss * Dd;        // 8.39M elems
    unsigned short* xb   = alloc(ND);
    unsigned short* yb   = alloc(ND);
    unsigned short* wkvb = alloc((size_t)Dd * 2 * Dd);
    unsigned short* wqb  = alloc((size_t)Dd * Dd);
    unsigned short* wob  = alloc((size_t)Dd * Dd);
    unsigned short* qb   = alloc(ND);
    unsigned short* kb   = alloc(ND);
    unsigned short* vb   = alloc(ND);
    unsigned short* ctxb = alloc(ND);

    auto cvt = [&](const float* src, unsigned short* dst, size_t n) {
        int n4 = (int)(n / 4);
        int blocks = (n4 + 255) / 256; if (blocks > 1024) blocks = 1024;
        cvt4_kernel<<<blocks, 256, 0, stream>>>((const float4*)src, (ushort4*)dst, n4);
    };
    cvt(x,   xb,   ND);
    cvt(y,   yb,   ND);
    cvt(Wkv, wkvb, (size_t)Dd * 2 * Dd);
    cvt(Wq,  wqb,  (size_t)Dd * Dd);
    cvt(Wo,  wob,  (size_t)Dd * Dd);

    const int M = Bb * Ss;   // 8192
    // kv = x @ Wkv + bkv   -> K,V [B,H,S,HD] bf16
    gemm_bf16_kernel<1><<<dim3((2 * Dd) / 128, M / 128), 256, 0, stream>>>(
        xb, wkvb, bkv, nullptr, kb, vb, M, 2 * Dd, Dd);
    // q = y @ Wq + bq      -> Q [B,H,S,HD] bf16
    gemm_bf16_kernel<2><<<dim3(Dd / 128, M / 128), 256, 0, stream>>>(
        yb, wqb, bq, nullptr, qb, nullptr, M, Dd, Dd);
    // attention -> ctx [B,S,D] bf16
    attn_kernel<<<dim3(Ss / 64, Bb * Hh), 128, 0, stream>>>(qb, kb, vb, mask, ctxb);
    // out = ctx @ Wo + bo  -> fp32 d_out
    gemm_bf16_kernel<0><<<dim3(Dd / 128, M / 128), 256, 0, stream>>>(
        ctxb, wob, bo, (float*)d_out, nullptr, nullptr, M, Dd, Dd);
}
